// NPTransitionPrior_7602092114230
// MI455X (gfx1250) — matrix operand order repaired and checked by reading, not compiled, not run on hardware
//
#include <hip/hip_runtime.h>
#include <hip/hip_bf16.h>

typedef __attribute__((ext_vector_type(16))) _Float16 v16h;
typedef __attribute__((ext_vector_type(8)))  float    v8f;

#define Dd     16
#define Hh     64
#define IN_F   33
#define Bb     32
#define Tt     512
#define NWIN   510            // T - L
#define Nn     (Bb * NWIN)    // 16320 rows
#define SLOPE  0.2f
#define TILES_PER_D (Nn / 16) // 1020 row-tiles per d
#define PITCH  72             // LDS row pitch in halves (16B-aligned A reads)

// packed f16 weights in d_ws (halves):
//   [layer(3)][d(16)][kstep(2)][jtile(4)][lane(32)][16]  (B-fragment layout)
#define WS_LAYER_STRIDE (Dd * 4096)          // 65536 halves / layer
#define WS_WOUT_OFF     (3 * WS_LAYER_STRIDE)// 196608
#define WS_TOTAL        (WS_WOUT_OFF + Dd * Hh) // 197632 halves (~386 KB)

// ---------------------------------------------------------------------------
// Pack weights into WMMA B-fragment layout; zero the log-det accumulators.
// B[k][h] = W[d][h][k]  (layer0 padded K: 33..63 -> 0)
// ---------------------------------------------------------------------------
__global__ void pack_kernel(const float* __restrict__ W0,
                            const float* __restrict__ W1,
                            const float* __restrict__ W2,
                            const float* __restrict__ Wout,
                            _Float16* __restrict__ wp,
                            float* __restrict__ outf) {
  int idx = blockIdx.x * blockDim.x + threadIdx.x;
  if (idx < Bb) outf[Nn * Dd + idx] = 0.0f;   // zero sum_log_abs_det accumulators
  if (idx >= WS_TOTAL) return;
  if (idx < WS_WOUT_OFF) {
    int layer = idx / WS_LAYER_STRIDE;
    int r = idx % WS_LAYER_STRIDE;
    int d = r / 4096; r %= 4096;
    int s = r / 2048; r %= 2048;
    int j = r / 512;  r %= 512;
    int lanek = r / 16;
    int e = r % 16;
    int k = 32 * s + (e < 8 ? 8 * (lanek / 16) + e : 16 + 8 * (lanek / 16) + (e - 8));      // B row (reduction index)
    int h = 16 * j + (lanek % 16);          // B column (output unit)
    float v;
    if (layer == 0)      v = (k < IN_F) ? W0[d * (Hh * IN_F) + h * IN_F + k] : 0.0f;
    else if (layer == 1) v = W1[d * (Hh * Hh) + h * Hh + k];
    else                 v = W2[d * (Hh * Hh) + h * Hh + k];
    wp[idx] = (_Float16)v;
  } else {
    int r = idx - WS_WOUT_OFF;
    int d = r / Hh, h = r % Hh;
    wp[idx] = (_Float16)Wout[d * Hh + h];  // Wout[d][0][h]
  }
}

// ---------------------------------------------------------------------------
// WMMA helpers
// ---------------------------------------------------------------------------
__device__ __forceinline__ v8f wmma_fma(v16h a, v16h b, v8f c) {
  // 8 args: (neg_a, A, neg_b, B, c_mod, C, reuse_a, reuse_b)
  return __builtin_amdgcn_wmma_f32_16x16x32_f16(false, a, false, b, (short)0, c,
                                                false, false);
}

__device__ __forceinline__ v16h loadW(const _Float16* __restrict__ wp,
                                      int layer, int d, int s, int j, int lane) {
  return *(const v16h*)(wp + (((layer * Dd + d) * 2 + s) * 2048 + j * 512 + lane * 16));
}

// A-fragment gather from a 16x64 f16 LDS tile (row m, 16-bit A layout):
//   lo lanes: K = kbase+{0..7}, kbase+{16..23};  hi lanes: +8 / +24
__device__ __forceinline__ v16h loadA(const _Float16* s, int m, int hi, int kbase) {
  const _Float16* p = s + m * PITCH + kbase + (hi ? 8 : 0);
  v16h r;
#pragma unroll
  for (int e = 0; e < 8; ++e) r[e] = p[e];
#pragma unroll
  for (int e = 0; e < 8; ++e) r[8 + e] = p[16 + e];
  return r;
}

// ---------------------------------------------------------------------------
// Main: one wave32 per (d, 16-row tile). Forward MLP + JVP via WMMA.
// ---------------------------------------------------------------------------
__global__ __launch_bounds__(128) void mlp_jvp_kernel(
    const float* __restrict__ x,
    const float* __restrict__ W0,
    const float* __restrict__ b0,
    const float* __restrict__ b1,
    const float* __restrict__ b2,
    const float* __restrict__ bout,
    const _Float16* __restrict__ wp,
    float* __restrict__ outf) {
  __shared__ _Float16 smF[4][16 * PITCH];  // forward activations (per wave)
  __shared__ _Float16 smT[4][16 * PITCH];  // tangent activations (per wave)

  const int wave = threadIdx.x >> 5;
  const int lane = threadIdx.x & 31;
  const int wid  = blockIdx.x * 4 + wave;
  const int dd   = wid / TILES_PER_D;      // which coordinate-MLP
  const int tile = wid % TILES_PER_D;      // which 16-row tile
  const int m    = lane & 15;              // A row held by this lane
  const int hi   = lane >> 4;
  const int n    = tile * 16 + m;          // global row
  const int bb   = n / NWIN;
  const int wdw  = n % NWIN;
  const float* xb = x + (bb * Tt + wdw) * Dd;  // window start: x[b][w][0]

  // ---- A fragments for layer 0 (16x64, two K-steps of 32) ----
  v16h a0, a1;
  const int offA = hi ? 8 : 0;
#pragma unroll
  for (int e = 0; e < 8; ++e) a0[e]     = (_Float16)xb[offA + e];        // K 0..7 / 8..15
#pragma unroll
  for (int e = 0; e < 8; ++e) a0[8 + e] = (_Float16)xb[16 + offA + e];   // K 16..23 / 24..31
#pragma unroll
  for (int e = 0; e < 16; ++e) a1[e] = (_Float16)0.0f;
  if (!hi) a1[0] = (_Float16)xb[2 * Dd + dd];  // K=32: xx = x[b][w+L][dd]

  v8f zero;
#pragma unroll
  for (int i = 0; i < 8; ++i) zero[i] = 0.0f;

  _Float16* sF = smF[wave];
  _Float16* sT = smT[wave];

  // ---- layer 0: pre0 = inp @ W0^T ----
  v8f c0[4];
#pragma unroll
  for (int j = 0; j < 4; ++j) {
    c0[j] = wmma_fma(a0, loadW(wp, 0, dd, 0, j, lane), zero);
    c0[j] = wmma_fma(a1, loadW(wp, 0, dd, 1, j, lane), c0[j]);
  }
  // bias + leaky, stage h0 and tangent seed t0 = g0 * W0[:,32]
#pragma unroll
  for (int j = 0; j < 4; ++j) {
    const int hcol  = 16 * j + m;                       // C layout: N = lane&15
    const float bv  = b0[dd * Hh + hcol];
    const float w32 = W0[dd * (Hh * IN_F) + hcol * IN_F + (IN_F - 1)];
#pragma unroll
    for (int r = 0; r < 8; ++r) {                       // C layout: M = r + 8*hi
      const float pre = c0[j][r] + bv;
      const float g   = pre > 0.0f ? 1.0f : SLOPE;
      const int row   = r + 8 * hi;
      sF[row * PITCH + hcol] = (_Float16)(g * pre);
      sT[row * PITCH + hcol] = (_Float16)(g * w32);
    }
  }
  __syncthreads();

  // ---- layers 1 and 2: forward and tangent GEMMs share B fragments ----
#pragma unroll
  for (int layer = 1; layer <= 2; ++layer) {
    v8f cF[4], cT[4];
#pragma unroll
    for (int j = 0; j < 4; ++j) { cF[j] = zero; cT[j] = zero; }
#pragma unroll
    for (int s = 0; s < 2; ++s) {
      v16h af = loadA(sF, m, hi, 32 * s);
      v16h at = loadA(sT, m, hi, 32 * s);
#pragma unroll
      for (int j = 0; j < 4; ++j) {
        v16h bw = loadW(wp, layer, dd, s, j, lane);
        cF[j] = wmma_fma(af, bw, cF[j]);
        cT[j] = wmma_fma(at, bw, cT[j]);
      }
    }
    __syncthreads();
    const float* bias = (layer == 1) ? b1 : b2;
#pragma unroll
    for (int j = 0; j < 4; ++j) {
      const int hcol = 16 * j + m;
      const float bv = bias[dd * Hh + hcol];
#pragma unroll
      for (int r = 0; r < 8; ++r) {
        const float pre = cF[j][r] + bv;
        const float g   = pre > 0.0f ? 1.0f : SLOPE;
        const int row   = r + 8 * hi;
        sF[row * PITCH + hcol] = (_Float16)(g * pre);
        sT[row * PITCH + hcol] = (_Float16)(g * cT[j][r]);
      }
    }
    __syncthreads();
  }

  // ---- output head: 64-wide dot for residual and Jacobian-diagonal ----
  if (lane < 16) {
    const _Float16* wo = wp + WS_WOUT_OFF + dd * Hh;
    const _Float16* rf = sF + m * PITCH;
    const _Float16* rt = sT + m * PITCH;
    float ov = 0.0f, dv = 0.0f;
#pragma unroll
    for (int h = 0; h < Hh; ++h) {
      const float wv = (float)wo[h];
      ov += wv * (float)rf[h];
      dv += wv * (float)rt[h];
    }
    ov += bout[dd];
    outf[n * Dd + dd] = ov;                              // residuals[b][w][d]
    atomicAdd(outf + Nn * Dd + bb, logf(fabsf(dv)));     // sum_log_abs_det[b]
  }
}

// ---------------------------------------------------------------------------
extern "C" void kernel_launch(void* const* d_in, const int* in_sizes, int n_in,
                              void* d_out, int out_size, void* d_ws, size_t ws_size,
                              hipStream_t stream) {
  (void)in_sizes; (void)n_in; (void)out_size; (void)ws_size;
  const float* x    = (const float*)d_in[0];
  const float* W0   = (const float*)d_in[1];
  const float* b0   = (const float*)d_in[2];
  const float* W1   = (const float*)d_in[3];
  const float* b1   = (const float*)d_in[4];
  const float* W2   = (const float*)d_in[5];
  const float* b2   = (const float*)d_in[6];
  const float* Wout = (const float*)d_in[7];
  const float* bout = (const float*)d_in[8];
  float*     outf = (float*)d_out;
  _Float16*  wp   = (_Float16*)d_ws;

  const int packThreads = 256;
  const int packBlocks  = (WS_TOTAL + packThreads - 1) / packThreads;
  pack_kernel<<<packBlocks, packThreads, 0, stream>>>(W0, W1, W2, Wout, wp, outf);

  const int blocks = (Dd * TILES_PER_D) / 4;  // 16320 waves / 4 per block = 4080
  mlp_jvp_kernel<<<blocks, 128, 0, stream>>>(x, W0, b0, b1, b2, bout, wp, outf);
}
